// ParallelSelfAttention_22342419874303
// MI455X (gfx1250) — matrix operand order repaired and checked by reading, not compiled, not run on hardware
//
#include <hip/hip_runtime.h>
#include <cmath>

#define S_LEN 4096
#define HID   2048
#define NHEAD 16
#define HDIM  128
#define QKV_N 6144   // 3*HID

typedef __attribute__((ext_vector_type(16))) __bf16 v16bf;
typedef __attribute__((ext_vector_type(8)))  float  v8f;
typedef unsigned short u16;
typedef unsigned int   u32;

__device__ __forceinline__ u16 f32_to_bf16(float f) {
  u32 u = __float_as_uint(f);
  u32 r = u + 0x7fffu + ((u >> 16) & 1u);   // round-to-nearest-even
  return (u16)(r >> 16);
}
__device__ __forceinline__ float bf16_to_f32(u16 h) {
  return __uint_as_float(((u32)h) << 16);
}
__device__ __forceinline__ v16bf make_frag(uint4 a, uint4 b) {
  union { uint4 q[2]; v16bf v; } u;
  u.q[0] = a; u.q[1] = b;
  return u.v;
}

// ---------------- f32 -> bf16 convert ----------------
__global__ void cvt_bf16_kernel(const float* __restrict__ in, u16* __restrict__ out, int n) {
  int i = blockIdx.x * blockDim.x + threadIdx.x;
  if (i < n) out[i] = f32_to_bf16(in[i]);
}

// ---------------- tiled WMMA GEMM: C[M][N] = A[M][K] * B[N][K]^T (+bias) ----------------
// 128x128 block tile, 256 threads = 8 waves (4 M x 2 N), wave computes 32x64.
// Double-buffered LDS, register prefetch, 1 barrier per K-step.
#define LDT 72   // LDS row stride in halves (144B: 16B aligned, conflict-free fragment gathers)
template <int N, int K, bool OUT_F32, bool HAS_BIAS>
__global__ __launch_bounds__(256) void gemm_bf16_kernel(
    const u16* __restrict__ A, const u16* __restrict__ B,
    const float* __restrict__ bias,
    float* __restrict__ Cf, u16* __restrict__ Cb)
{
  __shared__ u16 As[2][128 * LDT];
  __shared__ u16 Bs[2][128 * LDT];
  const int tid  = threadIdx.x;
  const int lane = tid & 31, wave = tid >> 5;
  const int wm = wave & 3, wn = wave >> 2;
  const int half = lane >> 4, l16 = lane & 15;
  const int bm = blockIdx.x * 128, bn = blockIdx.y * 128;

  // staging-chunk coordinates for this thread (2 chunks of 8 halves per array)
  const int r0 = tid >> 2,            kc0 = (tid & 3) * 8;
  const int r1 = (tid + 256) >> 2,    kc1 = ((tid + 256) & 3) * 8;

  v8f acc[2][4];
#pragma unroll
  for (int i = 0; i < 2; ++i)
#pragma unroll
    for (int j = 0; j < 4; ++j)
#pragma unroll
      for (int r = 0; r < 8; ++r) acc[i][j][r] = 0.0f;

  // prefetch tile 0 into registers
  uint4 pa0 = *(const uint4*)&A[(size_t)(bm + r0) * K + kc0];
  uint4 pa1 = *(const uint4*)&A[(size_t)(bm + r1) * K + kc1];
  uint4 pb0 = *(const uint4*)&B[(size_t)(bn + r0) * K + kc0];
  uint4 pb1 = *(const uint4*)&B[(size_t)(bn + r1) * K + kc1];

  int buf = 0;
  for (int k0 = 0; k0 < K; k0 += 32) {
    // commit staged registers to LDS[buf]
    *(uint4*)&As[buf][r0 * LDT + kc0] = pa0;
    *(uint4*)&As[buf][r1 * LDT + kc1] = pa1;
    *(uint4*)&Bs[buf][r0 * LDT + kc0] = pb0;
    *(uint4*)&Bs[buf][r1 * LDT + kc1] = pb1;
    __syncthreads();

    // prefetch next tile while computing
    if (k0 + 32 < K) {
      const int kn = k0 + 32;
      pa0 = *(const uint4*)&A[(size_t)(bm + r0) * K + kn + kc0];
      pa1 = *(const uint4*)&A[(size_t)(bm + r1) * K + kn + kc1];
      pb0 = *(const uint4*)&B[(size_t)(bn + r0) * K + kn + kc0];
      pb1 = *(const uint4*)&B[(size_t)(bn + r1) * K + kn + kc1];
    }

    v16bf af[2];
#pragma unroll
    for (int i = 0; i < 2; ++i) {
      const u16* ap = &As[buf][(wm * 32 + i * 16 + l16) * LDT + half * 8];
      af[i] = make_frag(*(const uint4*)ap, *(const uint4*)(ap + 16));
    }
#pragma unroll
    for (int j = 0; j < 4; ++j) {
      const u16* bp = &Bs[buf][(wn * 64 + j * 16 + l16) * LDT + half * 8];
      v16bf bf = make_frag(*(const uint4*)bp, *(const uint4*)(bp + 16));
#pragma unroll
      for (int i = 0; i < 2; ++i)
        acc[i][j] = __builtin_amdgcn_wmma_f32_16x16x32_bf16(
            false, af[i], false, bf, (short)0, acc[i][j], false, false);
    }
    buf ^= 1;
  }

  // epilogue: C layout row = r + 8*half, col = l16 (per 16x16 tile); base ptr + r*N offsets
#pragma unroll
  for (int i = 0; i < 2; ++i)
#pragma unroll
    for (int j = 0; j < 4; ++j) {
      const int col  = bn + wn * 64 + j * 16 + l16;
      const int row0 = bm + wm * 32 + i * 16 + 8 * half;
      float bv = 0.0f;
      if (HAS_BIAS) bv = bias[col];
      if (OUT_F32) {
        float* p = Cf + (size_t)row0 * N + col;
#pragma unroll
        for (int r = 0; r < 8; ++r) p[r * N] = acc[i][j][r] + bv;
      } else {
        u16* p = Cb + (size_t)row0 * N + col;
#pragma unroll
        for (int r = 0; r < 8; ++r) p[r * N] = f32_to_bf16(acc[i][j][r] + bv);
      }
    }
}

// ---------------- RoPE (first 32 dims, NeoX rotate-half) + repack ----------------
// qkv[s][head*384 + {q:0..127, k:128..255, v:256..383}]
// -> Q,K: [head][s][128] bf16 ; V transposed: [head][128][s] bf16
__global__ void rope_repack_kernel(const u16* __restrict__ qkv,
                                   u16* __restrict__ Qo, u16* __restrict__ Ko,
                                   u16* __restrict__ Vo)
{
  int idx = blockIdx.x * blockDim.x + threadIdx.x;
  if (idx >= S_LEN * NHEAD * HDIM) return;
  int hn   = idx & (HDIM - 1);
  int head = (idx >> 7) & (NHEAD - 1);
  int s    = idx >> 11;
  const u16* row = qkv + (size_t)s * QKV_N + head * 384;
  float qv = bf16_to_f32(row[hn]);
  float kv = bf16_to_f32(row[HDIM + hn]);
  if (hn < 32) {
    int j = hn & 15;                                  // freq index (emb = concat(freqs,freqs))
    float inv_freq = __expf(-0.57564627f * (float)j); // 10000^(-j/16)
    float f = (float)s * inv_freq;
    float sn, cs; __sincosf(f, &sn, &cs);
    int other = (hn < 16) ? hn + 16 : hn - 16;
    float sgn = (hn < 16) ? -1.0f : 1.0f;             // rot = concat(-x2, x1)
    qv = qv * cs + sgn * bf16_to_f32(row[other]) * sn;
    kv = kv * cs + sgn * bf16_to_f32(row[HDIM + other]) * sn;
  }
  size_t qk = ((size_t)head * S_LEN + s) * HDIM + hn;
  Qo[qk] = f32_to_bf16(qv);
  Ko[qk] = f32_to_bf16(kv);
  Vo[((size_t)head * HDIM + hn) * S_LEN + s] = row[256 + hn];  // raw bf16 copy, transposed
}

// ---------------- flash attention (causal, scale 1/sqrt(128)) ----------------
// grid (S/128, NHEAD); 8 waves; wave w owns queries q0+16w..+15, full 128-dim O accum.
#define FA_KT 64
#define KS_LD 136
#define VT_LD 72
#define P_LD  72
__global__ __launch_bounds__(256) void flash_attn_kernel(
    const u16* __restrict__ Q, const u16* __restrict__ Kk, const u16* __restrict__ Vt,
    u16* __restrict__ ctx, float scale)
{
  __shared__ u16 Ks[FA_KT * KS_LD];
  __shared__ u16 Vs[HDIM * VT_LD];
  __shared__ u16 Ps[8 * 16 * P_LD];

  const int tid  = threadIdx.x;
  const int lane = tid & 31, wave = tid >> 5;
  const int half = lane >> 4, l16 = lane & 15;
  const int head = blockIdx.y;
  const int q0   = blockIdx.x * 128;

  const u16* Qh = Q  + (size_t)head * S_LEN * HDIM;
  const u16* Kh = Kk + (size_t)head * S_LEN * HDIM;
  const u16* Vh = Vt + (size_t)head * HDIM * S_LEN;

  // Q fragments for this wave's 16 rows (hn split into 4 K-steps of 32)
  v16bf qf[4];
  const u16* qp = Qh + (size_t)(q0 + wave * 16 + l16) * HDIM;
#pragma unroll
  for (int ks = 0; ks < 4; ++ks) {
    int kb = ks * 32 + half * 8;
    qf[ks] = make_frag(*(const uint4*)(qp + kb), *(const uint4*)(qp + kb + 16));
  }

  v8f oacc[8];
  float mrow[8], lrow[8];
#pragma unroll
  for (int t = 0; t < 8; ++t) {
#pragma unroll
    for (int r = 0; r < 8; ++r) oacc[t][r] = 0.0f;
    mrow[t] = -1.0e30f; lrow[t] = 0.0f;
  }

  const int ntiles = q0 / FA_KT + 2;               // causal bound for this block
  for (int t = 0; t < ntiles; ++t) {
    const int kt0 = t * FA_KT;
    __syncthreads();                               // previous P·V reads done
    // stage K tile [64][128] and V^T tile [128][64]; 1024 16B-chunks each
#pragma unroll
    for (int c = 0; c < 4; ++c) {
      int ch  = tid + c * 256;
      int kr  = ch >> 4, kc = (ch & 15) * 8;
      *(uint4*)&Ks[kr * KS_LD + kc] = *(const uint4*)&Kh[(size_t)(kt0 + kr) * HDIM + kc];
      int vr  = ch >> 3, vc = (ch & 7) * 8;
      *(uint4*)&Vs[vr * VT_LD + vc] = *(const uint4*)&Vh[(size_t)vr * S_LEN + kt0 + vc];
    }
    __syncthreads();

    // scores: 4 N-tiles of 16 keys x 4 hn K-steps
    v8f sacc[4];
#pragma unroll
    for (int nt = 0; nt < 4; ++nt) {
#pragma unroll
      for (int r = 0; r < 8; ++r) sacc[nt][r] = 0.0f;
#pragma unroll
      for (int ks = 0; ks < 4; ++ks) {
        const u16* bp = &Ks[(nt * 16 + l16) * KS_LD + ks * 32 + half * 8];
        v16bf bf = make_frag(*(const uint4*)bp, *(const uint4*)(bp + 16));
        sacc[nt] = __builtin_amdgcn_wmma_f32_16x16x32_bf16(
            false, qf[ks], false, bf, (short)0, sacc[nt], false, false);
      }
    }

    // scale + causal mask (-10000, matching reference)
    const bool need_mask = (kt0 + FA_KT - 1) > (q0 + wave * 16);
#pragma unroll
    for (int nt = 0; nt < 4; ++nt)
#pragma unroll
      for (int r = 0; r < 8; ++r) {
        float v = sacc[nt][r] * scale;
        if (need_mask) {
          int key = kt0 + nt * 16 + l16;
          int qr  = q0 + wave * 16 + r + 8 * half;
          if (key > qr) v = -10000.0f;
        }
        sacc[nt][r] = v;
      }

    // online softmax (row lives in 16 lanes of a half; reduce with shfl_xor 1/2/4/8)
#pragma unroll
    for (int r = 0; r < 8; ++r) {
      float rmax = fmaxf(fmaxf(sacc[0][r], sacc[1][r]), fmaxf(sacc[2][r], sacc[3][r]));
#pragma unroll
      for (int d = 1; d < 16; d <<= 1) rmax = fmaxf(rmax, __shfl_xor(rmax, d, 32));
      float mnew  = fmaxf(mrow[r], rmax);
      float alpha = __expf(mrow[r] - mnew);
      mrow[r] = mnew;
      float rs = 0.0f;
#pragma unroll
      for (int nt = 0; nt < 4; ++nt) {
        float p = __expf(sacc[nt][r] - mnew);
        sacc[nt][r] = p;
        rs += p;
      }
#pragma unroll
      for (int d = 1; d < 16; d <<= 1) rs += __shfl_xor(rs, d, 32);
      lrow[r] = lrow[r] * alpha + rs;
#pragma unroll
      for (int nt = 0; nt < 8; ++nt) oacc[nt][r] *= alpha;
    }

    // C-layout -> A-layout reshape of P through LDS (bf16)
    u16* pw = &Ps[wave * 16 * P_LD];
#pragma unroll
    for (int nt = 0; nt < 4; ++nt)
#pragma unroll
      for (int r = 0; r < 8; ++r)
        pw[(r + 8 * half) * P_LD + nt * 16 + l16] = f32_to_bf16(sacc[nt][r]);
    __syncthreads();

    // O += P (16x64) * V (64x128): 2 K-steps x 8 hn tiles
#pragma unroll
    for (int ks = 0; ks < 2; ++ks) {
      const u16* ap = &pw[l16 * P_LD + ks * 32 + half * 8];
      v16bf pf = make_frag(*(const uint4*)ap, *(const uint4*)(ap + 16));
#pragma unroll
      for (int nt = 0; nt < 8; ++nt) {
        const u16* vp = &Vs[(nt * 16 + l16) * VT_LD + ks * 32 + half * 8];
        v16bf vf = make_frag(*(const uint4*)vp, *(const uint4*)(vp + 16));
        oacc[nt] = __builtin_amdgcn_wmma_f32_16x16x32_bf16(
            false, pf, false, vf, (short)0, oacc[nt], false, false);
      }
    }
  }

  // normalize and store ctx[s][head*128 + hn] as bf16
#pragma unroll
  for (int r = 0; r < 8; ++r) {
    float inv = 1.0f / lrow[r];
    int qr = q0 + wave * 16 + r + 8 * half;
    u16* cp = ctx + (size_t)qr * HID + head * HDIM + l16;
#pragma unroll
    for (int nt = 0; nt < 8; ++nt)
      cp[nt * 16] = f32_to_bf16(oacc[nt][r] * inv);
  }
}

// ---------------- launch ----------------
extern "C" void kernel_launch(void* const* d_in, const int* in_sizes, int n_in,
                              void* d_out, int out_size, void* d_ws, size_t ws_size,
                              hipStream_t stream)
{
  const float* hidden  = (const float*)d_in[0];
  (void)d_in[1];                              // attention_mask == causal triu, computed from indices
  const float* w_qkv   = (const float*)d_in[2];
  const float* w_dense = (const float*)d_in[3];
  const float* b_dense = (const float*)d_in[4];
  (void)in_sizes; (void)n_in; (void)out_size; (void)ws_size;

  u16* ws      = (u16*)d_ws;
  u16* hid_b   = ws;                                   // 4096*2048
  u16* wqkv_b  = hid_b   + (size_t)S_LEN * HID;        // 6144*2048
  u16* wdens_b = wqkv_b  + (size_t)QKV_N * HID;        // 2048*2048
  u16* qkv_t   = wdens_b + (size_t)HID * HID;          // 4096*6144
  u16* Qb      = qkv_t   + (size_t)S_LEN * QKV_N;      // 16*4096*128
  u16* Kb      = Qb      + (size_t)NHEAD * S_LEN * HDIM;
  u16* Vb      = Kb      + (size_t)NHEAD * S_LEN * HDIM;
  u16* ctx     = Vb      + (size_t)NHEAD * S_LEN * HDIM;

  const int n1 = S_LEN * HID, n2 = QKV_N * HID, n3 = HID * HID;
  cvt_bf16_kernel<<<(n1 + 255) / 256, 256, 0, stream>>>(hidden,  hid_b,   n1);
  cvt_bf16_kernel<<<(n2 + 255) / 256, 256, 0, stream>>>(w_qkv,   wqkv_b,  n2);
  cvt_bf16_kernel<<<(n3 + 255) / 256, 256, 0, stream>>>(w_dense, wdens_b, n3);

  gemm_bf16_kernel<QKV_N, HID, false, false>
      <<<dim3(S_LEN / 128, QKV_N / 128), 256, 0, stream>>>(
          hid_b, wqkv_b, nullptr, nullptr, qkv_t);

  const int nr = S_LEN * NHEAD * HDIM;
  rope_repack_kernel<<<(nr + 255) / 256, 256, 0, stream>>>(qkv_t, Qb, Kb, Vb);

  flash_attn_kernel<<<dim3(S_LEN / 128, NHEAD), 256, 0, stream>>>(
      Qb, Kb, Vb, ctx, 1.0f / sqrtf((float)HDIM));

  gemm_bf16_kernel<HID, HID, true, true>
      <<<dim3(S_LEN / 128, HID / 128), 256, 0, stream>>>(
          ctx, wdens_b, b_dense, (float*)d_out, nullptr);
}
